// DeepARS4_74010876445305
// MI455X (gfx1250) — compile-verified, matrix operand order checked by reading, not run on hardware
//
#include <hip/hip_runtime.h>
#include <hip/hip_bf16.h>
#include <stdint.h>

// ---------------------------------------------------------------------------
// DeepAR-S4D forward for MI455X (gfx1250, wave32).
//   B=32, L=2048, D_IN=33, H=512, NL=4, N2=32
// S4D FFT conv replaced by exact diagonal SSM recurrence (same math, ZOH).
// Dominant cost: per-layer GEMM (65536 x 512) @ (512 x 1024) -> bf16 WMMA.
// ---------------------------------------------------------------------------

#define Bsz   32
#define Lsz   2048
#define DIN   33
#define Hsz   512
#define NLAY  4
#define NMODE 32
#define Msz   (Bsz * Lsz)          // 65536 rows

typedef __attribute__((ext_vector_type(16))) __bf16 v16bf;
typedef __attribute__((ext_vector_type(8)))  __bf16 v8bf;
typedef __attribute__((ext_vector_type(8)))  float  v8f;
typedef __attribute__((ext_vector_type(4)))  uint32_t v4u;
typedef __attribute__((ext_vector_type(8)))  int      v8i;
typedef __attribute__((ext_vector_type(4)))  int      v4i;

// ------------------------- device helpers ---------------------------------

__device__ __forceinline__ float gelu_exact(float x) {
  return 0.5f * x * (1.0f + erff(x * 0.70710678118654752f));
}
__device__ __forceinline__ float sigmoidf(float x) {
  return 1.0f / (1.0f + __expf(-x));
}
__device__ __forceinline__ float softplusf(float x) {
  return (x > 20.0f) ? x : log1pf(__expf(x));
}

#if __has_builtin(__builtin_amdgcn_tensor_load_to_lds) && __has_builtin(__builtin_amdgcn_s_wait_tensorcnt)
#define USE_TDM 1
#else
#define USE_TDM 0
#endif

#if USE_TDM
// Issue one 2D TDM tile load: global (row-major, 2-byte elems) -> LDS.
// D# packing per CDNA5 ISA ch.8 (group0: flags/lds/global addr; group1: dims).
// This toolchain exposes the 6-arg builtin: (g0 v4u, g1 v8i, g2 v4i, g3 v4i,
// extra v8i, cpol i32); groups 2/3 and the extra group are unused for 2D.
__device__ __forceinline__ void tdm_load_2d(uint32_t lds_off, const void* gptr,
                                            uint32_t tensor_d0, uint32_t tensor_d1,
                                            uint32_t tile_d0, uint32_t tile_d1,
                                            uint32_t stride0) {
  uint64_t ga = (uint64_t)(uintptr_t)gptr;
  v4u g0;
  g0[0] = 1u;                                            // count=1, user mode
  g0[1] = lds_off;                                       // LDS byte address
  g0[2] = (uint32_t)ga;                                  // global_addr[31:0]
  g0[3] = (uint32_t)((ga >> 32) & 0x1FFFFFFu) | (2u << 30); // addr[56:32] | type=2
  v8i g1;
  g1[0] = (int)(1u << 16);                               // data_size=1 (2 bytes)
  g1[1] = (int)((tensor_d0 & 0xFFFFu) << 16);            // tensor_dim0[15:0] @ bit48
  g1[2] = (int)(((tensor_d0 >> 16) & 0xFFFFu) | ((tensor_d1 & 0xFFFFu) << 16));
  g1[3] = (int)(((tensor_d1 >> 16) & 0xFFFFu) | (tile_d0 << 16));
  g1[4] = (int)(tile_d1 & 0xFFFFu);                      // tile_dim1, tile_dim2=0
  g1[5] = (int)stride0;                                  // tensor_dim0_stride[31:0]
  g1[6] = 0;
  g1[7] = 0;
  v4i gz4 = {0, 0, 0, 0};
  v8i gz8 = {0, 0, 0, 0, 0, 0, 0, 0};
  __builtin_amdgcn_tensor_load_to_lds(g0, g1, gz4, gz4, gz8, 0);
}
#endif

// Load a 16x32 bf16 WMMA fragment from an LDS tile stored row-major with
// 32 bf16 per row. Matches documented A layout: lanes 0-15 hold K 0..7,16..23;
// lanes 16-31 hold K 8..15,24..31 (B fragment assumed mirrored, col-per-lane).
__device__ __forceinline__ v16bf load_frag(const __bf16* lds, int row, int sel) {
  const v8bf* p0 = (const v8bf*)(lds + row * 32 + sel * 8);
  const v8bf* p1 = (const v8bf*)(lds + row * 32 + sel * 8 + 16);
  v8bf lo = *p0, hi = *p1;
  return __builtin_shufflevector(lo, hi, 0, 1, 2, 3, 4, 5, 6, 7,
                                 8, 9, 10, 11, 12, 13, 14, 15);
}

// ------------------------- kernels -----------------------------------------

// out_w (NL, 2H, H) f32 -> bf16 (one pass, reused by all 4 layer GEMMs)
__global__ __launch_bounds__(256) void wconv_kernel(const float* __restrict__ w,
                                                    __bf16* __restrict__ o, int n) {
  int i = blockIdx.x * 256 + threadIdx.x;
  if (i < n) o[i] = (__bf16)w[i];
}

// h = x @ enc_w + enc_b   (K=33 is tiny: VALU FMA with x row staged in LDS)
__global__ __launch_bounds__(256) void encoder_kernel(const float* __restrict__ x,
                                                      const float* __restrict__ enc_w,
                                                      const float* __restrict__ enc_b,
                                                      float* __restrict__ hbuf) {
  __shared__ float xs[DIN];
  int m = blockIdx.x;
  int t = threadIdx.x;
  if (t < DIN) xs[t] = x[(size_t)m * DIN + t];
  __syncthreads();
  for (int n = t; n < Hsz; n += 256) {
    float acc = enc_b[n];
#pragma unroll
    for (int k = 0; k < DIN; ++k) acc += xs[k] * enc_w[k * Hsz + n];
    hbuf[(size_t)m * Hsz + n] = acc;
  }
}

// S4D diagonal SSM scan + D-skip + exact GELU, emitting bf16 activations.
// One lane per (b, h); 32 complex modes live in VGPRs; u reads / y writes are
// coalesced across the 32 lanes (lane = channel).
__global__ __launch_bounds__(32) void scan_kernel(const float* __restrict__ hbuf,
                                                  const float* __restrict__ log_dt,
                                                  const float* __restrict__ Cp,
                                                  const float* __restrict__ log_A_real,
                                                  const float* __restrict__ A_imag,
                                                  const float* __restrict__ Dp,
                                                  __bf16* __restrict__ yact,
                                                  int layer) {
  int hh = blockIdx.x * 32 + threadIdx.x;
  int b  = blockIdx.y;
  float dt   = __expf(log_dt[layer * Hsz + hh]);
  float dval = Dp[layer * Hsz + hh];

  float wre[NMODE], wim[NMODE], c2re[NMODE], c2im[NMODE], sre[NMODE], sim[NMODE];
#pragma unroll
  for (int n = 0; n < NMODE; ++n) {
    int pid = (layer * Hsz + hh) * NMODE + n;
    float Are = -__expf(log_A_real[pid]);
    float Aim = A_imag[pid];
    float dr = Are * dt, di = Aim * dt;
    float er = __expf(dr);
    float cw_re = er * __cosf(di), cw_im = er * __sinf(di);  // w = exp(dt*A)
    float Ere = cw_re - 1.0f, Eim = cw_im;                   // exp(dtA) - 1
    float inv = 1.0f / (Are * Are + Aim * Aim);
    float qre = (Ere * Are + Eim * Aim) * inv;               // (exp(dtA)-1)/A
    float qim = (Eim * Are - Ere * Aim) * inv;
    float Cre = Cp[pid * 2 + 0], Cim = Cp[pid * 2 + 1];
    wre[n] = cw_re; wim[n] = cw_im;
    c2re[n] = 2.0f * (Cre * qre - Cim * qim);                // 2*CC
    c2im[n] = 2.0f * (Cre * qim + Cim * qre);
    sre[n] = 0.0f; sim[n] = 0.0f;
  }

  size_t idx = ((size_t)b * Lsz) * Hsz + hh;
  for (int l = 0; l < Lsz; ++l, idx += Hsz) {
    float u = hbuf[idx];
    float acc = dval * u;
#pragma unroll
    for (int n = 0; n < NMODE; ++n) {
      float nr = wre[n] * sre[n] - wim[n] * sim[n] + u;      // s = w*s + u
      float ni = wre[n] * sim[n] + wim[n] * sre[n];
      sre[n] = nr; sim[n] = ni;
      acc += c2re[n] * nr - c2im[n] * ni;                    // 2*Re(CC*s)
    }
    yact[idx] = (__bf16)gelu_exact(acc);
  }
}

// z = GLU( yact @ W^T + b )  via bf16 WMMA, f32 accumulate.
// Block tile: M=128 rows, 64 'a' cols paired with 64 'g' cols (same GLU gate).
// A tile (128x32) and W tiles (64x32 + 64x32) staged in LDS by the TDM.
__global__ __launch_bounds__(256) void gemmglu_kernel(const __bf16* __restrict__ yact,
                                                      const __bf16* __restrict__ wbf,
                                                      const float* __restrict__ out_b,
                                                      float* __restrict__ zbuf,
                                                      int layer) {
  __shared__ __bf16 ldsA[128 * 32];
  __shared__ __bf16 ldsB[128 * 32];   // rows 0..63 = 'a' weights, 64..127 = 'g'
  const int n0 = blockIdx.x * 64;
  const int m0 = blockIdx.y * 128;
  const __bf16* Wl = wbf + (size_t)layer * (2 * Hsz) * Hsz;
  const int lane = threadIdx.x & 31;
  const int wave = threadIdx.x >> 5;
  const int sel  = lane >> 4;

  v8f zero = {0.f, 0.f, 0.f, 0.f, 0.f, 0.f, 0.f, 0.f};
  v8f acc_a[4], acc_g[4];
#pragma unroll
  for (int j = 0; j < 4; ++j) { acc_a[j] = zero; acc_g[j] = zero; }

#if USE_TDM
  const uint32_t ldsA_off = (uint32_t)(uintptr_t)(void*)ldsA;
  const uint32_t ldsB_off = (uint32_t)(uintptr_t)(void*)ldsB;
#endif

  for (int k0 = 0; k0 < Hsz; k0 += 32) {
    __syncthreads();   // previous iteration's LDS fully consumed
#if USE_TDM
    if (threadIdx.x < 32) {   // wave 0 drives the Tensor Data Mover
      tdm_load_2d(ldsA_off, yact + (size_t)m0 * Hsz + k0,
                  Hsz, Msz, 32, 128, Hsz);
      tdm_load_2d(ldsB_off, Wl + (size_t)n0 * Hsz + k0,
                  Hsz, 2 * Hsz, 32, 64, Hsz);
      tdm_load_2d(ldsB_off + 64 * 32 * 2, Wl + (size_t)(Hsz + n0) * Hsz + k0,
                  Hsz, 2 * Hsz, 32, 64, Hsz);
      __builtin_amdgcn_s_wait_tensorcnt(0);
    }
#else
    for (int e = threadIdx.x; e < 128 * 32; e += 256) {
      int rr = e >> 5, kk = e & 31;
      ldsA[e] = yact[(size_t)(m0 + rr) * Hsz + k0 + kk];
      int wr = (rr < 64) ? (n0 + rr) : (Hsz + n0 + rr - 64);
      ldsB[e] = Wl[(size_t)wr * Hsz + k0 + kk];
    }
#endif
    __syncthreads();

    v16bf af = load_frag(ldsA, wave * 16 + (lane & 15), sel);
#pragma unroll
    for (int j = 0; j < 4; ++j) {
      v16bf bfa = load_frag(ldsB, j * 16 + (lane & 15), sel);
      acc_a[j] = __builtin_amdgcn_wmma_f32_16x16x32_bf16(
          false, af, false, bfa, (short)0, acc_a[j], false, false);
      v16bf bfg = load_frag(ldsB, 64 + j * 16 + (lane & 15), sel);
      acc_g[j] = __builtin_amdgcn_wmma_f32_16x16x32_bf16(
          false, af, false, bfg, (short)0, acc_g[j], false, false);
    }
  }

  // Epilogue: bias + GLU, scatter per documented C layout
  // (VGPR v, lane l) -> M = v + 8*(l>=16), N = l&15.
#pragma unroll
  for (int j = 0; j < 4; ++j) {
    int col = n0 + j * 16 + (lane & 15);
    float ba = out_b[layer * (2 * Hsz) + col];
    float bg = out_b[layer * (2 * Hsz) + Hsz + col];
#pragma unroll
    for (int v = 0; v < 8; ++v) {
      int mrow = m0 + wave * 16 + v + (sel << 3);
      float a = acc_a[j][v] + ba;
      float g = acc_g[j][v] + bg;
      zbuf[(size_t)mrow * Hsz + col] = a * sigmoidf(g);
    }
  }
}

// h = LayerNorm(z + h) * w + b  (one block per (b,l) row)
__global__ __launch_bounds__(256) void ln_kernel(const float* __restrict__ zbuf,
                                                 float* __restrict__ hbuf,
                                                 const float* __restrict__ norm_w,
                                                 const float* __restrict__ norm_b,
                                                 int layer) {
  __shared__ float red[256];
  size_t m = blockIdx.x;
  int t = threadIdx.x;
  float v0 = zbuf[m * Hsz + t]       + hbuf[m * Hsz + t];
  float v1 = zbuf[m * Hsz + t + 256] + hbuf[m * Hsz + t + 256];
  red[t] = v0 + v1;
  __syncthreads();
  for (int s = 128; s > 0; s >>= 1) {
    if (t < s) red[t] += red[t + s];
    __syncthreads();
  }
  float mu = red[0] * (1.0f / (float)Hsz);
  __syncthreads();
  float d0 = v0 - mu, d1 = v1 - mu;
  red[t] = d0 * d0 + d1 * d1;
  __syncthreads();
  for (int s = 128; s > 0; s >>= 1) {
    if (t < s) red[t] += red[t + s];
    __syncthreads();
  }
  float inv = rsqrtf(red[0] * (1.0f / (float)Hsz) + 1e-5f);
  hbuf[m * Hsz + t]       = d0 * inv * norm_w[layer * Hsz + t]       + norm_b[layer * Hsz + t];
  hbuf[m * Hsz + t + 256] = d1 * inv * norm_w[layer * Hsz + t + 256] + norm_b[layer * Hsz + t + 256];
}

// mu/alpha = softplus(h @ w + b)  — one wave per row, shfl_xor reduction
__global__ __launch_bounds__(256) void heads_kernel(const float* __restrict__ hbuf,
                                                    const float* __restrict__ mu_w,
                                                    const float* __restrict__ mu_b,
                                                    const float* __restrict__ al_w,
                                                    const float* __restrict__ al_b,
                                                    float* __restrict__ out) {
  int wave = threadIdx.x >> 5;
  int lane = threadIdx.x & 31;
  size_t m = (size_t)blockIdx.x * 8 + wave;
  float am = 0.f, aa = 0.f;
#pragma unroll
  for (int e = 0; e < Hsz / 32; ++e) {
    float hv = hbuf[m * Hsz + lane + 32 * e];
    am += hv * mu_w[lane + 32 * e];
    aa += hv * al_w[lane + 32 * e];
  }
  for (int off = 16; off > 0; off >>= 1) {
    am += __shfl_xor(am, off, 32);
    aa += __shfl_xor(aa, off, 32);
  }
  if (lane == 0) {
    out[m]       = softplusf(am + mu_b[0]);
    out[Msz + m] = softplusf(aa + al_b[0]);
  }
}

// ------------------------- host launcher -----------------------------------

extern "C" void kernel_launch(void* const* d_in, const int* in_sizes, int n_in,
                              void* d_out, int out_size, void* d_ws, size_t ws_size,
                              hipStream_t stream) {
  const float* x          = (const float*)d_in[0];
  const float* enc_w      = (const float*)d_in[1];
  const float* enc_b      = (const float*)d_in[2];
  const float* log_dt     = (const float*)d_in[3];
  const float* Cp         = (const float*)d_in[4];
  const float* log_A_real = (const float*)d_in[5];
  const float* A_imag     = (const float*)d_in[6];
  const float* Dp         = (const float*)d_in[7];
  const float* out_w      = (const float*)d_in[8];
  const float* out_b      = (const float*)d_in[9];
  const float* norm_w     = (const float*)d_in[10];
  const float* norm_b     = (const float*)d_in[11];
  const float* mu_w       = (const float*)d_in[12];
  const float* mu_b       = (const float*)d_in[13];
  const float* alpha_w    = (const float*)d_in[14];
  const float* alpha_b    = (const float*)d_in[15];
  float* out = (float*)d_out;

  // Workspace layout (bytes):
  //   h     f32 : M*H*4   = 134217728
  //   yact  bf16: M*H*2   =  67108864
  //   z     f32 : M*H*4   = 134217728
  //   Wbf16     : NL*2H*H*2 = 4194304      total ~324 MB
  char* ws = (char*)d_ws;
  float*  hbuf = (float*)(ws);
  __bf16* yact = (__bf16*)(ws + 134217728ull);
  float*  zbuf = (float*)(ws + 201326592ull);
  __bf16* wbf  = (__bf16*)(ws + 335544320ull);

  const int wn = NLAY * 2 * Hsz * Hsz;
  wconv_kernel<<<(wn + 255) / 256, 256, 0, stream>>>(out_w, wbf, wn);
  encoder_kernel<<<Msz, 256, 0, stream>>>(x, enc_w, enc_b, hbuf);

  for (int i = 0; i < NLAY; ++i) {
    scan_kernel<<<dim3(Hsz / 32, Bsz), 32, 0, stream>>>(
        hbuf, log_dt, Cp, log_A_real, A_imag, Dp, yact, i);
    gemmglu_kernel<<<dim3(Hsz / 64, Msz / 128), 256, 0, stream>>>(
        yact, wbf, out_b, zbuf, i);
    ln_kernel<<<Msz, 256, 0, stream>>>(zbuf, hbuf, norm_w, norm_b, i);
  }

  heads_kernel<<<Msz / 8, 256, 0, stream>>>(hbuf, mu_w, mu_b, alpha_w, alpha_b, out);
}
